// PixelwiseFlowPredictor_48498770706582
// MI455X (gfx1250) — compile-verified, matrix operand order checked by reading, not run on hardware
//
#include <hip/hip_runtime.h>
#include <hip/hip_bf16.h>

// ---------------------------------------------------------------------------
// Problem constants (match the reference)
// ---------------------------------------------------------------------------
constexpr int Bc  = 4;
constexpr int NRc = 10;
constexpr int Cc  = 3;
constexpr int Hc  = 256;
constexpr int Wc  = 256;
constexpr int HWc = Hc * Wc;
constexpr int SMPLD = 170;
constexpr int OUTF  = 108;             // 64 + (NR+1)*(C+1)

// d_out layout (flat, in return order, fp32)
constexpr size_t FLOW_OFF = 0;                         // (B,H,W,2)
constexpr size_t COMB_OFF = (size_t)Bc * HWc * 2;      // (B,1,H,W)
constexpr size_t SM_OFF   = COMB_OFF + (size_t)Bc * HWc;
constexpr size_t OCC_OFF  = SM_OFF   + (size_t)Bc * HWc;

typedef __attribute__((ext_vector_type(16))) _Float16 v16h;
typedef __attribute__((ext_vector_type(8)))  _Float16 v8h;
typedef __attribute__((ext_vector_type(8)))  float    v8f;
typedef __attribute__((ext_vector_type(4)))  int      v4i;

#define USE_ASYNC_LDS 1

// 16B global -> LDS copy; async (ASYNCcnt, no VGPR round trip) when available
__device__ __forceinline__ void stage_b128(const _Float16* g, _Float16* l)
{
#if defined(USE_ASYNC_LDS) && __has_builtin(__builtin_amdgcn_global_load_async_to_lds_b128)
    typedef __attribute__((address_space(1))) v4i* gp_t;
    typedef __attribute__((address_space(3))) v4i* lp_t;
    __builtin_amdgcn_global_load_async_to_lds_b128((gp_t)g, (lp_t)l, 0, 0);
#else
    *(v8h*)l = *(const v8h*)g;
#endif
}

__device__ __forceinline__ void stage_wait()
{
#if defined(USE_ASYNC_LDS) && __has_builtin(__builtin_amdgcn_global_load_async_to_lds_b128)
#if __has_builtin(__builtin_amdgcn_s_wait_asynccnt)
    __builtin_amdgcn_s_wait_asynccnt(0);
#else
    asm volatile("s_wait_asynccnt 0" ::: "memory");
#endif
#endif
}

// WMMA fragment per ISA 7.12.2 (lanes 0-15: K0-7/16-23; 16-31: K8-15/24-31)
__device__ __forceinline__ v16h frag16(const _Float16* row, int kc, int kb)
{
    const v8h lo = *(const v8h*)(row + kc + kb * 8);
    const v8h hi = *(const v8h*)(row + kc + 16 + kb * 8);
    v16h f;
#pragma unroll
    for (int i = 0; i < 8; ++i) { f[i] = lo[i]; f[i + 8] = hi[i]; }
    return f;
}

// ---------------------------------------------------------------------------
// 1) prep: heatmaps + sparse motions + 11-region bilinear warp of source
// ---------------------------------------------------------------------------
__global__ void prep_k(const float* __restrict__ img,
                       const float* __restrict__ dshift,
                       const float* __restrict__ sshift,
                       const float* __restrict__ daff,
                       const float* __restrict__ saff,
                       float* __restrict__ sparse,
                       _Float16* __restrict__ pred_in)
{
    size_t idx = blockIdx.x * (size_t)blockDim.x + threadIdx.x;
    if (idx >= (size_t)Bc * HWc) return;
    int b = (int)(idx / HWc);
    int pix = (int)(idx % HWc);
    int y = pix / Wc, x = pix % Wc;
    float gx = 2.f * x / (Wc - 1.f) - 1.f;
    float gy = 2.f * y / (Hc - 1.f) - 1.f;

    float sxk[NRc + 1], syk[NRc + 1], heatk[NRc + 1];
    sxk[0] = gx; syk[0] = gy; heatk[0] = 0.f;

    for (int r = 0; r < NRc; ++r) {
        const float* ds = &dshift[((size_t)b * NRc + r) * 2];
        const float* ss = &sshift[((size_t)b * NRc + r) * 2];
        float zx = gx - ds[0], zy = gy - ds[1];
        float gd = __expf(-0.5f * (zx * zx + zy * zy) / 0.01f);
        float ux = gx - ss[0], uy = gy - ss[1];
        float gs = __expf(-0.5f * (ux * ux + uy * uy) / 0.01f);
        heatk[r + 1] = gd - gs;

        const float* A = &daff[((size_t)b * NRc + r) * 4];
        const float* S = &saff[((size_t)b * NRc + r) * 4];
        float det  = A[0] * A[3] - A[1] * A[2];
        float rdet = 1.f / det;
        float i00 =  A[3] * rdet, i01 = -A[1] * rdet;
        float i10 = -A[2] * rdet, i11 =  A[0] * rdet;
        float m00 = S[0] * i00 + S[1] * i10;
        float m01 = S[0] * i01 + S[1] * i11;
        float m10 = S[2] * i00 + S[3] * i10;
        float m11 = S[2] * i01 + S[3] * i11;
        sxk[r + 1] = m00 * zx + m01 * zy + ss[0];
        syk[r + 1] = m10 * zx + m11 * zy + ss[1];
    }

    const float* im = img + (size_t)b * Cc * HWc;
    for (int k = 0; k < NRc + 1; ++k) {
        float sx = sxk[k], sy = syk[k];
        size_t so = (((size_t)b * (NRc + 1) + k) * HWc + pix) * 2;
        sparse[so + 0] = sx;
        sparse[so + 1] = sy;
        float fx = (sx + 1.f) * (Wc * 0.5f) - 0.5f;
        float fy = (sy + 1.f) * (Hc * 0.5f) - 0.5f;
        float x0f = floorf(fx), y0f = floorf(fy);
        int x0 = (int)x0f, y0 = (int)y0f;
        float wx = fx - x0f, wy = fy - y0f;
        float rgb[3] = {0.f, 0.f, 0.f};
        for (int dy = 0; dy < 2; ++dy)
            for (int dx = 0; dx < 2; ++dx) {
                int ix = x0 + dx, iy = y0 + dy;
                if (ix < 0 || ix >= Wc || iy < 0 || iy >= Hc) continue;
                float wgt = (dx ? wx : 1.f - wx) * (dy ? wy : 1.f - wy);
                for (int c = 0; c < Cc; ++c)
                    rgb[c] += wgt * im[(size_t)c * HWc + (size_t)iy * Wc + ix];
            }
        _Float16* pb = pred_in + ((size_t)b * 44 + (size_t)k * 4) * HWc + pix;
        pb[0]               = (_Float16)heatk[k];
        pb[(size_t)HWc]     = (_Float16)rgb[0];
        pb[(size_t)2 * HWc] = (_Float16)rgb[1];
        pb[(size_t)3 * HWc] = (_Float16)rgb[2];
    }
}

// ---------------------------------------------------------------------------
// 2) weight repack: fp32 (Cout,Cin,ks,ks) -> f16 N-major (Np x Kp), zero pad
// ---------------------------------------------------------------------------
__global__ void wprep_k(const float* __restrict__ w, _Float16* __restrict__ Wt,
                        int Cout, int Cin, int ks, int Kp, int Np)
{
    size_t idx = blockIdx.x * (size_t)blockDim.x + threadIdx.x;
    size_t total = (size_t)Np * Kp;
    if (idx >= total) return;
    int n = (int)(idx / Kp);
    int k = (int)(idx % Kp);
    int ks2 = ks * ks;
    int K = Cin * ks2;
    float v = 0.f;
    if (n < Cout && k < K) {
        int cin = k / ks2, rem = k % ks2;
        v = w[(((size_t)n * Cin + cin) * ks + rem / ks) * ks + rem % ks];
    }
    Wt[idx] = (_Float16)v;
}

// ---------------------------------------------------------------------------
// 3) implicit-GEMM conv, WMMA f16->f32
//    Block tile: 64(M) x 64(N), K staged 64 per iteration.
//    4 waves (2x2); wave = 32x32 C tile = 2x2 accumulators (frag reuse x2).
// ---------------------------------------------------------------------------
constexpr int KC = 64;                 // K per LDS stage

__global__ __launch_bounds__(128)
void conv_wmma_k(const _Float16* __restrict__ in0, int Cin0,
                 const _Float16* __restrict__ in1, int Cin1,
                 const _Float16* __restrict__ Wt, int Kp,
                 const float* __restrict__ bias,
                 const float* __restrict__ style,
                 _Float16* __restrict__ out, int Cout,
                 int Hin, int Win, int Hout, int Wout,
                 int ks, int pad, int up, int relu)
{
    __shared__ alignas(16) _Float16 As[64][KC + 8];   // [m][k]
    __shared__ alignas(16) _Float16 Bs[64][KC + 8];   // [n][k] (N-major like Wt)

    const int b    = blockIdx.z;
    const int mt   = blockIdx.x * 64;
    const int nt   = blockIdx.y * 64;
    const int t    = threadIdx.x;
    const int lane = t & 31;
    const int wid  = t >> 5;
    const int mo   = (wid & 1) * 32;   // wave M offset (two 16-row subtiles)
    const int no   = (wid >> 1) * 32;  // wave N offset (two 16-col subtiles)

    const int ks2  = ks * ks;
    const int CinT = Cin0 + Cin1;
    const size_t HWin  = (size_t)Hin * Win;
    const size_t HWout = (size_t)Hout * Wout;
    const _Float16* base0 = in0 + (size_t)b * Cin0 * HWin;
    const _Float16* base1 = in1 ? (in1 + (size_t)b * Cin1 * HWin) : (const _Float16*)0;

    // A staging map: 128 threads cover 64 rows x 64 k (32 k-elems per thread)
    const int arow = t >> 1;           // 0..63
    const int akb  = (t & 1) * 32;     // 0 or 32
    const int m    = mt + arow;
    const int oy   = m / Wout, ox = m % Wout;
    // B staging map: 64 rows x 64 k (four b128 per thread)
    const int bn  = t >> 1;            // 0..63
    const int bkb = (t & 1) * 32;      // 0 or 32

    v8f acc00 = {}, acc01 = {}, acc10 = {}, acc11 = {};
    for (int k0 = 0; k0 < Kp; k0 += KC) {
        // ---- stage B (weights): four async 16B copies per thread ----
        {
            const _Float16* g = &Wt[(size_t)(nt + bn) * Kp + (k0 + bkb)];
            _Float16* l = &Bs[bn][bkb];
            stage_b128(g,      l);
            stage_b128(g + 8,  l + 8);
            stage_b128(g + 16, l + 16);
            stage_b128(g + 24, l + 24);
        }
        // ---- stage A (im2col, division-free incremental k-walk) ----
        {
            int kg  = k0 + akb;
            int cin = kg / ks2;
            int rem = kg - cin * ks2;
            int kh  = rem / ks;
            int kw  = rem - kh * ks;
#pragma unroll 4
            for (int i = 0; i < 32; ++i) {
                float v = 0.f;
                if (cin < CinT) {
                    int yy = oy - pad + kh;
                    int xx = ox - pad + kw;
                    if (yy >= 0 && yy < Hout && xx >= 0 && xx < Wout) {
                        int sy = up ? (yy >> 1) : yy;
                        int sx = up ? (xx >> 1) : xx;
                        const _Float16* p = (cin < Cin0)
                            ? &base0[(size_t)cin * HWin + (size_t)sy * Win + sx]
                            : &base1[(size_t)(cin - Cin0) * HWin + (size_t)sy * Win + sx];
                        v = (float)*p;
                        if (style) v *= style[(size_t)b * CinT + cin];
                    }
                }
                As[arow][akb + i] = (_Float16)v;
                if (++kw == ks) { kw = 0; if (++kh == ks) { kh = 0; ++cin; } }
            }
        }
        stage_wait();
        __syncthreads();

        const int fr = lane & 15, kb = lane >> 4;
        const _Float16* arow0 = &As[mo + fr][0];
        const _Float16* arow1 = &As[mo + 16 + fr][0];
        const _Float16* brow0 = &Bs[no + fr][0];
        const _Float16* brow1 = &Bs[no + 16 + fr][0];
#pragma unroll
        for (int kc = 0; kc < KC; kc += 32) {
            const v16h a0 = frag16(arow0, kc, kb);
            const v16h a1 = frag16(arow1, kc, kb);
            const v16h b0 = frag16(brow0, kc, kb);
            const v16h b1 = frag16(brow1, kc, kb);
            acc00 = __builtin_amdgcn_wmma_f32_16x16x32_f16(
                        false, a0, false, b0, (short)0, acc00, false, false);
            acc01 = __builtin_amdgcn_wmma_f32_16x16x32_f16(
                        false, a0, false, b1, (short)0, acc01, false, false);
            acc10 = __builtin_amdgcn_wmma_f32_16x16x32_f16(
                        false, a1, false, b0, (short)0, acc10, false, false);
            acc11 = __builtin_amdgcn_wmma_f32_16x16x32_f16(
                        false, a1, false, b1, (short)0, acc11, false, false);
        }
        __syncthreads();
    }

    // ---- epilogue: bias + optional ReLU, f16 stores (C/D layout 7.12.2) ----
    const int fr = lane & 15;
    const int hi = (lane >> 4) ? 8 : 0;
    const int n0 = nt + no + fr;
    const int n1 = nt + no + 16 + fr;
    const int mb0 = mt + mo + hi;
    const int mb1 = mt + mo + 16 + hi;
    if (n0 < Cout) {
        const float bv = bias ? bias[n0] : 0.f;
        _Float16* ob = out + ((size_t)b * Cout + n0) * HWout;
#pragma unroll
        for (int j = 0; j < 8; ++j) {
            float v0 = acc00[j] + bv;
            float v1 = acc10[j] + bv;
            if (relu) { v0 = fmaxf(v0, 0.f); v1 = fmaxf(v1, 0.f); }
            ob[mb0 + j] = (_Float16)v0;
            ob[mb1 + j] = (_Float16)v1;
        }
    }
    if (n1 < Cout) {
        const float bv = bias ? bias[n1] : 0.f;
        _Float16* ob = out + ((size_t)b * Cout + n1) * HWout;
#pragma unroll
        for (int j = 0; j < 8; ++j) {
            float v0 = acc01[j] + bv;
            float v1 = acc11[j] + bv;
            if (relu) { v0 = fmaxf(v0, 0.f); v1 = fmaxf(v1, 0.f); }
            ob[mb0 + j] = (_Float16)v0;
            ob[mb1 + j] = (_Float16)v1;
        }
    }
}

// ---------------------------------------------------------------------------
// 4) 2x2 average pool (f16)
// ---------------------------------------------------------------------------
__global__ void avgpool2_k(const _Float16* __restrict__ in,
                           _Float16* __restrict__ out, int BC, int Hi, int Wi)
{
    int Ho = Hi >> 1, Wo = Wi >> 1;
    size_t idx = blockIdx.x * (size_t)blockDim.x + threadIdx.x;
    if (idx >= (size_t)BC * Ho * Wo) return;
    int bc = (int)(idx / ((size_t)Ho * Wo));
    int r  = (int)(idx % ((size_t)Ho * Wo));
    int y = r / Wo, x = r % Wo;
    const _Float16* p = in + (size_t)bc * Hi * Wi + (size_t)(2 * y) * Wi + 2 * x;
    float s = (float)p[0] + (float)p[1] + (float)p[Wi] + (float)p[Wi + 1];
    out[idx] = (_Float16)(0.25f * s);
}

// ---------------------------------------------------------------------------
// 5) style = smpl @ mod_w.T + mod_b
// ---------------------------------------------------------------------------
__global__ void style_k(const float* __restrict__ dsm, const float* __restrict__ ssm,
                        const float* __restrict__ mw, const float* __restrict__ mb,
                        float* __restrict__ style)
{
    int idx = blockIdx.x * blockDim.x + threadIdx.x;
    if (idx >= Bc * OUTF) return;
    int b = idx / OUTF, o = idx % OUTF;
    float s = mb[o];
    for (int j = 0; j < SMPLD; ++j) {
        float sv = (j < 85) ? dsm[b * 85 + j] : ssm[b * 85 + (j - 85)];
        s += sv * mw[(size_t)o * SMPLD + j];
    }
    style[idx] = s;
}

// ---------------------------------------------------------------------------
// 6) epilogue part 1: softmax(mask) -> deformation; smpl flow*mask; occlusion
// ---------------------------------------------------------------------------
__global__ void final_k(const _Float16* __restrict__ maskL,
                        const _Float16* __restrict__ occL,
                        const float* __restrict__ sparse,
                        const float* __restrict__ sflow,
                        const float* __restrict__ smask,
                        float* __restrict__ comb_in,
                        float* __restrict__ outp)
{
    size_t idx = blockIdx.x * (size_t)blockDim.x + threadIdx.x;
    if (idx >= (size_t)Bc * HWc) return;
    int b = (int)(idx / HWc);
    int pix = (int)(idx % HWc);

    float lg[NRc + 1], mx = -1e30f;
    for (int k = 0; k <= NRc; ++k) {
        lg[k] = (float)maskL[((size_t)b * (NRc + 1) + k) * HWc + pix];
        mx = fmaxf(mx, lg[k]);
    }
    float sum = 0.f;
    for (int k = 0; k <= NRc; ++k) { lg[k] = __expf(lg[k] - mx); sum += lg[k]; }
    float inv = 1.f / sum;
    float dx = 0.f, dy = 0.f;
    for (int k = 0; k <= NRc; ++k) {
        float mk = lg[k] * inv;
        size_t so = (((size_t)b * (NRc + 1) + k) * HWc + pix) * 2;
        dx += mk * sparse[so + 0];
        dy += mk * sparse[so + 1];
    }
    float sm = smask[(size_t)b * HWc + pix];   // bilinear resize to same size == identity
    float fx = sflow[((size_t)b * HWc + pix) * 2 + 0] * sm;
    float fy = sflow[((size_t)b * HWc + pix) * 2 + 1] * sm;

    float* cb = comb_in + (size_t)b * 5 * HWc + pix;
    cb[0]               = sm;
    cb[(size_t)HWc]     = dx;
    cb[(size_t)2 * HWc] = dy;
    cb[(size_t)3 * HWc] = fx;
    cb[(size_t)4 * HWc] = fy;

    float o = (float)occL[(size_t)b * HWc + pix];
    outp[OCC_OFF + (size_t)b * HWc + pix] = 1.f / (1.f + __expf(-o));
}

// ---------------------------------------------------------------------------
// 7) epilogue part 2: 5->1 3x3 combine conv + flow blend + outputs
// ---------------------------------------------------------------------------
__global__ void comb_k(const float* __restrict__ comb_in,
                       const float* __restrict__ cw, const float* __restrict__ cb,
                       float* __restrict__ outp)
{
    size_t idx = blockIdx.x * (size_t)blockDim.x + threadIdx.x;
    if (idx >= (size_t)Bc * HWc) return;
    int b = (int)(idx / HWc);
    int pix = (int)(idx % HWc);
    int y = pix / Wc, x = pix % Wc;

    const float* ci = comb_in + (size_t)b * 5 * HWc;
    float acc = cb[0];
    for (int c = 0; c < 5; ++c)
        for (int ky = 0; ky < 3; ++ky)
            for (int kx = 0; kx < 3; ++kx) {
                int yy = y - 1 + ky, xx = x - 1 + kx;
                if (yy < 0 || yy >= Hc || xx < 0 || xx >= Wc) continue;
                acc += cw[((size_t)c * 3 + ky) * 3 + kx] *
                       ci[(size_t)c * HWc + (size_t)yy * Wc + xx];
            }
    float comb = acc;
    float sm = ci[pix];
    float dx = ci[(size_t)HWc + pix],     dy = ci[(size_t)2 * HWc + pix];
    float fx = ci[(size_t)3 * HWc + pix], fy = ci[(size_t)4 * HWc + pix];
    float ofx = dx * (1.f - comb) + fx * comb;
    float ofy = dy * (1.f - comb) + fy * comb;

    outp[FLOW_OFF + ((size_t)b * HWc + pix) * 2 + 0] = ofx;
    outp[FLOW_OFF + ((size_t)b * HWc + pix) * 2 + 1] = ofy;
    outp[COMB_OFF + (size_t)b * HWc + pix] = comb;
    outp[SM_OFF   + (size_t)b * HWc + pix] = sm;
}

// ---------------------------------------------------------------------------
// host
// ---------------------------------------------------------------------------
extern "C" void kernel_launch(void* const* d_in, const int* in_sizes, int n_in,
                              void* d_out, int out_size, void* d_ws, size_t ws_size,
                              hipStream_t stream)
{
    (void)in_sizes; (void)n_in; (void)out_size; (void)ws_size;

    const float* src_img = (const float*)d_in[0];
    const float* dshift  = (const float*)d_in[1];
    const float* sshift  = (const float*)d_in[2];
    const float* daff    = (const float*)d_in[3];
    const float* saff    = (const float*)d_in[4];
    const float* dsmpl   = (const float*)d_in[5];
    const float* ssmpl   = (const float*)d_in[6];
    const float* sflow   = (const float*)d_in[7];
    const float* smask   = (const float*)d_in[8];
    const float* ew[5], *eb[5], *dw[5], *db[5];
    for (int i = 0; i < 5; ++i) { ew[i] = (const float*)d_in[9 + 2 * i];  eb[i] = (const float*)d_in[10 + 2 * i]; }
    for (int i = 0; i < 5; ++i) { dw[i] = (const float*)d_in[19 + 2 * i]; db[i] = (const float*)d_in[20 + 2 * i]; }
    const float* mask_mw = (const float*)d_in[29];
    const float* mask_mb = (const float*)d_in[30];
    const float* mask_w  = (const float*)d_in[31];
    const float* mask_b  = (const float*)d_in[32];
    const float* occ_mw  = (const float*)d_in[33];
    const float* occ_mb  = (const float*)d_in[34];
    const float* occ_w   = (const float*)d_in[35];
    const float* occ_b   = (const float*)d_in[36];
    const float* comb_w  = (const float*)d_in[37];
    const float* comb_b  = (const float*)d_in[38];
    float* outp = (float*)d_out;

    char* ws = (char*)d_ws;
    size_t off = 0;
    auto alloc = [&](size_t bytes) -> void* {
        void* p = ws + off;
        off = (off + bytes + 255) & ~(size_t)255;
        return p;
    };
    float*     sparse  = (float*)    alloc((size_t)Bc * 11 * HWc * 2 * 4);
    _Float16*  pred_in = (_Float16*) alloc((size_t)Bc * 44 * HWc * 2);
    _Float16*  cbuf    = (_Float16*) alloc((size_t)Bc * 128 * HWc * 2);
    _Float16*  p0 = (_Float16*)alloc((size_t)Bc * 128 * 128 * 128 * 2);
    _Float16*  p1 = (_Float16*)alloc((size_t)Bc * 256 * 64 * 64 * 2);
    _Float16*  p2 = (_Float16*)alloc((size_t)Bc * 512 * 32 * 32 * 2);
    _Float16*  p3 = (_Float16*)alloc((size_t)Bc * 1024 * 16 * 16 * 2);
    _Float16*  p4 = (_Float16*)alloc((size_t)Bc * 1024 * 8 * 8 * 2);
    _Float16*  t0 = (_Float16*)alloc((size_t)Bc * 1024 * 16 * 16 * 2);
    _Float16*  t1 = (_Float16*)alloc((size_t)Bc * 512 * 32 * 32 * 2);
    _Float16*  t2 = (_Float16*)alloc((size_t)Bc * 256 * 64 * 64 * 2);
    _Float16*  t3 = (_Float16*)alloc((size_t)Bc * 128 * 128 * 128 * 2);
    _Float16*  t4 = (_Float16*)alloc((size_t)Bc * 64 * HWc * 2);
    _Float16*  maskL = (_Float16*)alloc((size_t)Bc * 11 * HWc * 2);
    _Float16*  occL  = (_Float16*)alloc((size_t)Bc * 1 * HWc * 2);
    float*     comb_in = (float*)alloc((size_t)Bc * 5 * HWc * 4);
    float*     styleM  = (float*)alloc((size_t)Bc * OUTF * 4);
    float*     styleO  = (float*)alloc((size_t)Bc * OUTF * 4);
    _Float16*  WtBuf   = (_Float16*)alloc((size_t)9437184 * 2);  // largest Np*Kp (512x18432)

    {
        size_t n = (size_t)Bc * HWc;
        prep_k<<<(unsigned)((n + 255) / 256), 256, 0, stream>>>(
            src_img, dshift, sshift, daff, saff, sparse, pred_in);
        style_k<<<(Bc * OUTF + 255) / 256, 256, 0, stream>>>(dsmpl, ssmpl, mask_mw, mask_mb, styleM);
        style_k<<<(Bc * OUTF + 255) / 256, 256, 0, stream>>>(dsmpl, ssmpl, occ_mw, occ_mb, styleO);
    }

    auto run_conv = [&](const _Float16* in0, int Cin0, const _Float16* in1, int Cin1,
                        const float* w, const float* bias, const float* styleP,
                        _Float16* outb, int Cout, int Hin, int Win, int Hout, int Wout,
                        int ks, int pad, int up, int relu) {
        int CinT = Cin0 + Cin1;
        int K  = CinT * ks * ks;
        int Kp = ((K + KC - 1) / KC) * KC;           // pad K to stage size
        int Np = ((Cout + 63) / 64) * 64;            // pad N to block tile
        size_t total = (size_t)Np * Kp;
        wprep_k<<<(unsigned)((total + 255) / 256), 256, 0, stream>>>(w, WtBuf, Cout, CinT, ks, Kp, Np);
        dim3 grid((unsigned)((Hout * Wout) / 64), (unsigned)(Np / 64), (unsigned)Bc);
        conv_wmma_k<<<grid, 128, 0, stream>>>(in0, Cin0, in1, Cin1, WtBuf, Kp, bias, styleP,
                                              outb, Cout, Hin, Win, Hout, Wout, ks, pad, up, relu);
    };
    auto run_pool = [&](const _Float16* in, _Float16* outb, int Ch, int Hi) {
        size_t n = (size_t)Bc * Ch * (Hi / 2) * (Hi / 2);
        avgpool2_k<<<(unsigned)((n + 255) / 256), 256, 0, stream>>>(in, outb, Bc * Ch, Hi, Hi);
    };

    // encoder
    run_conv(pred_in, 44,  nullptr, 0, ew[0], eb[0], nullptr, cbuf, 128, 256, 256, 256, 256, 3, 1, 0, 1);
    run_pool(cbuf, p0, 128, 256);
    run_conv(p0, 128, nullptr, 0, ew[1], eb[1], nullptr, cbuf, 256, 128, 128, 128, 128, 3, 1, 0, 1);
    run_pool(cbuf, p1, 256, 128);
    run_conv(p1, 256, nullptr, 0, ew[2], eb[2], nullptr, cbuf, 512, 64, 64, 64, 64, 3, 1, 0, 1);
    run_pool(cbuf, p2, 512, 64);
    run_conv(p2, 512, nullptr, 0, ew[3], eb[3], nullptr, cbuf, 1024, 32, 32, 32, 32, 3, 1, 0, 1);
    run_pool(cbuf, p3, 1024, 32);
    run_conv(p3, 1024, nullptr, 0, ew[4], eb[4], nullptr, cbuf, 1024, 16, 16, 16, 16, 3, 1, 0, 1);
    run_pool(cbuf, p4, 1024, 16);

    // decoder (upsample folded; skip-concat folded into two-pointer input)
    run_conv(p4, 1024, nullptr, 0, dw[0], db[0], nullptr, t0, 1024, 8, 8, 16, 16, 3, 1, 1, 1);
    run_conv(t0, 1024, p3, 1024, dw[1], db[1], nullptr, t1, 512, 16, 16, 32, 32, 3, 1, 1, 1);
    run_conv(t1, 512,  p2, 512,  dw[2], db[2], nullptr, t2, 256, 32, 32, 64, 64, 3, 1, 1, 1);
    run_conv(t2, 256,  p1, 256,  dw[3], db[3], nullptr, t3, 128, 64, 64, 128, 128, 3, 1, 1, 1);
    run_conv(t3, 128,  p0, 128,  dw[4], db[4], nullptr, t4, 64, 128, 128, 256, 256, 3, 1, 1, 1);

    // style-modulated 7x7 convs on prediction = [t4(64) | pred_in(44)]
    run_conv(t4, 64, pred_in, 44, mask_w, mask_b, styleM, maskL, 11, 256, 256, 256, 256, 7, 3, 0, 0);
    run_conv(t4, 64, pred_in, 44, occ_w,  occ_b,  styleO, occL,   1, 256, 256, 256, 256, 7, 3, 0, 0);

    // epilogue
    {
        size_t n = (size_t)Bc * HWc;
        final_k<<<(unsigned)((n + 255) / 256), 256, 0, stream>>>(
            maskL, occL, sparse, sflow, smask, comb_in, outp);
        comb_k<<<(unsigned)((n + 255) / 256), 256, 0, stream>>>(comb_in, comb_w, comb_b, outp);
    }
}
